// AGCRN_79706003079778
// MI455X (gfx1250) — matrix operand order repaired and checked by reading, not compiled, not run on hardware
//
#include <hip/hip_runtime.h>
#include <hip/hip_bf16.h>

// ---------------------------------------------------------------------------
// AGCRN on gfx1250: fp16 WMMA (v_wmma_f32_16x16x32_f16) for the graph
// propagation GEMM (A @ concat(x,h)) and per-node adaptive GEMMs.
// All fragments load as contiguous 16B vectors (ds_load_b128 /
// global_load_b128); staging loops have static trip counts; B fragments are
// batch-loaded so WMMAs issue back-to-back.
// ---------------------------------------------------------------------------

typedef __attribute__((ext_vector_type(16))) _Float16 v16h;
typedef __attribute__((ext_vector_type(8)))  _Float16 v8h;
typedef __attribute__((ext_vector_type(8)))  float    v8f;

#define N_NODES 512
#define BATCH   64
#define TSTEPS  24
#define CIN     2
#define HID     64
#define DEMB    16
#define IGATE   66          // C + H
#define KI      132         // K * IGATE
#define KPAD    160         // KI padded to multiple of 32
#define IPAD    80          // per-batch column pad (66 -> 80, multiple of 16)
#define XCOLS   (BATCH * IPAD)   // 5120 columns in the X / AX matrices
#define XGS     168         // LDS stride (halves) for the 64 x 160 xg tile
#define TS      40          // LDS stride (halves) for 32-wide k tiles

__device__ inline v16h combine8(v8h lo, v8h hi) {
    return __builtin_shufflevector(lo, hi, 0, 1, 2, 3, 4, 5, 6, 7,
                                   8, 9, 10, 11, 12, 13, 14, 15);
}

// A fragment (16x32 f16): lane l -> row m=l%16, half=l/16.
// Element j -> k = 16*half + j (matches B fragments):
// 16 contiguous halves -> 2x ds_load_b128 from a row-major tile.
__device__ inline v16h frag_a(const _Float16* tile, int stride, int row0, int k0) {
    int lane = threadIdx.x & 31;
    int m = (lane & 15) + row0, hf = lane >> 4;
    const _Float16* p = tile + m * stride + k0 + 16 * hf;
    v8h lo = *(const v8h*)p;
    v8h hi = *(const v8h*)(p + 8);
    return combine8(lo, hi);
}

// B fragment (32x16 f16): lane l -> col n=l%16, k-group l/16; element j ->
// k = (l/16)*16 + j : 16 contiguous k per lane -> 2x b128 from a
// column-major (transposed) tile.
__device__ inline v16h frag_b_t(const _Float16* tileT, int stride, int col0) {
    int lane = threadIdx.x & 31;
    const _Float16* p = tileT + (col0 + (lane & 15)) * stride + (lane >> 4) * 16;
    v8h lo = *(const v8h*)p;
    v8h hi = *(const v8h*)(p + 8);
    return combine8(lo, hi);
}

// B fragment straight from global node-weights stored (n, o, k) with k-stride
// KPAD: 16 contiguous halves per lane -> 2x global_load_b128.
__device__ inline v16h frag_b_w(const _Float16* Wn_n, int col0, int k0) {
    int lane = threadIdx.x & 31;
    const _Float16* p = Wn_n + (size_t)(col0 + (lane & 15)) * KPAD + k0 + (lane >> 4) * 16;
    v8h lo = *(const v8h*)p;
    v8h hi = *(const v8h*)(p + 8);
    return combine8(lo, hi);
}

__device__ inline v8f wmma_f16(v16h a, v16h b, v8f c) {
    return __builtin_amdgcn_wmma_f32_16x16x32_f16(false, a, false, b, (short)0, c,
                                                  false, false);
}

// -------------------- setup kernel: adjacency softmax ----------------------
__global__ void adj_kernel(const float* __restrict__ E1, _Float16* __restrict__ A16) {
    __shared__ float row[N_NODES];
    __shared__ float red[128];
    __shared__ float en[DEMB];
    int n = blockIdx.x, tid = threadIdx.x;
    if (tid < DEMB) en[tid] = E1[n * DEMB + tid];
    __syncthreads();
    for (int m = tid; m < N_NODES; m += 128) {
        float acc = 0.f;
#pragma unroll
        for (int d = 0; d < DEMB; ++d) acc += en[d] * E1[m * DEMB + d];
        row[m] = fmaxf(acc, 0.f);
    }
    __syncthreads();
    float mx = -1e30f;
    for (int m = tid; m < N_NODES; m += 128) mx = fmaxf(mx, row[m]);
    red[tid] = mx;
    __syncthreads();
    for (int s = 64; s > 0; s >>= 1) {
        if (tid < s) red[tid] = fmaxf(red[tid], red[tid + s]);
        __syncthreads();
    }
    float mxv = red[0];
    __syncthreads();
    float sm = 0.f;
    for (int m = tid; m < N_NODES; m += 128) {
        float e = __expf(row[m] - mxv);
        row[m] = e;
        sm += e;
    }
    red[tid] = sm;
    __syncthreads();
    for (int s = 64; s > 0; s >>= 1) {
        if (tid < s) red[tid] += red[tid + s];
        __syncthreads();
    }
    float inv = 1.f / red[0];
    for (int m = tid; m < N_NODES; m += 128)
        A16[(size_t)n * N_NODES + m] = (_Float16)(row[m] * inv);
}

// ----- setup: node-adaptive weights, (n, o, k) layout, zero K-padded -------
__global__ void nodew_kernel(const float* __restrict__ E1,
                             const float* __restrict__ Wpool,
                             _Float16* __restrict__ Wn, int Og) {
    size_t idx = (size_t)blockIdx.x * blockDim.x + threadIdx.x;
    int ki = (int)(idx % KPAD);
    int o  = (int)((idx / KPAD) % Og);
    int n  = (int)(idx / ((size_t)Og * KPAD));
    float a = 0.f;
    if (ki < KI) {
        int k = ki / IGATE, i = ki % IGATE;
#pragma unroll
        for (int d = 0; d < DEMB; ++d)
            a += E1[n * DEMB + d] * Wpool[(((size_t)d * 2 + k) * IGATE + i) * Og + o];
    }
    Wn[idx] = (_Float16)a;
}

// ------------- setup kernel: node-adaptive biases --------------------------
__global__ void bias_kernel(const float* __restrict__ E1,
                            const float* __restrict__ bzr,
                            const float* __restrict__ bc,
                            float* __restrict__ bnzr, float* __restrict__ bnc) {
    int idx = blockIdx.x * blockDim.x + threadIdx.x;
    if (idx < N_NODES * 128) {
        int n = idx >> 7, o = idx & 127;
        float a = 0.f;
#pragma unroll
        for (int d = 0; d < DEMB; ++d) a += E1[n * DEMB + d] * bzr[d * 128 + o];
        bnzr[idx] = a;
    } else {
        int j = idx - N_NODES * 128;
        int n = j >> 6, o = j & 63;
        float a = 0.f;
#pragma unroll
        for (int d = 0; d < DEMB; ++d) a += E1[n * DEMB + d] * bc[d * 64 + o];
        bnc[j] = a;
    }
}

// ----------------- per-step: write x_t feature columns ---------------------
__global__ void pack_kernel(const float* __restrict__ x, int t,
                            _Float16* __restrict__ X1, _Float16* __restrict__ X2) {
    int idx = blockIdx.x * blockDim.x + threadIdx.x;   // B*N*C
    int c = idx & 1;
    int n = (idx >> 1) & (N_NODES - 1);
    int b = idx >> 10;
    float v = x[(((size_t)b * TSTEPS + t) * N_NODES + n) * CIN + c];
    size_t dst = (size_t)n * XCOLS + b * IPAD + c;
    X1[dst] = (_Float16)v;
    X2[dst] = (_Float16)v;
}

// --------------- graph propagation GEMM: AX = A (512x512) @ X (512x5120) ---
// 64x64 macro tile, 4 waves/block. X tile staged transposed so that B
// fragments are 2x ds_load_b128; all 4 B frags loaded before the 4 WMMAs.
__global__ void gemmA_kernel(const _Float16* __restrict__ A16,
                             const _Float16* __restrict__ Xin,
                             _Float16* __restrict__ AXout) {
    __shared__ _Float16 Atile[64 * TS];      // [row m][k], stride 40
    __shared__ _Float16 XtileT[64 * TS];     // [col n][k], stride 40 (transposed)
    int tid = threadIdx.x;
    int wave = tid >> 5;
    int mbase = blockIdx.x * 64;
    int nbase = blockIdx.y * 64;

    v8f acc[4] = {};
    for (int kc = 0; kc < N_NODES / 32; ++kc) {
        int kbase = kc * 32;
        __syncthreads();
        // A tile: 64 x 32 halves, 2 vector copies per thread (static trip count)
#pragma unroll
        for (int u = 0; u < 2; ++u) {
            int i = tid + u * 128;
            int h0 = i * 8, r = h0 >> 5, c = h0 & 31;
            *(uint4*)&Atile[r * TS + c] =
                *(const uint4*)&A16[(size_t)(mbase + r) * N_NODES + kbase + c];
        }
        // X tile: 32(k) x 64(n) in memory -> transposed scatter into LDS
#pragma unroll
        for (int u = 0; u < 2; ++u) {
            int i = tid + u * 128;
            int h0 = i * 8, r = h0 >> 6, c = h0 & 63;   // r = k row, c = n col
            union { uint4 u4; _Float16 h[8]; } tmp;
            tmp.u4 = *(const uint4*)&Xin[(size_t)(kbase + r) * XCOLS + nbase + c];
#pragma unroll
            for (int q = 0; q < 8; ++q) XtileT[(c + q) * TS + r] = tmp.h[q];
        }
        __syncthreads();
        v16h af  = frag_a(Atile, TS, wave * 16, 0);
        v16h bf0 = frag_b_t(XtileT, TS, 0);
        v16h bf1 = frag_b_t(XtileT, TS, 16);
        v16h bf2 = frag_b_t(XtileT, TS, 32);
        v16h bf3 = frag_b_t(XtileT, TS, 48);
        acc[0] = wmma_f16(af, bf0, acc[0]);
        acc[1] = wmma_f16(af, bf1, acc[1]);
        acc[2] = wmma_f16(af, bf2, acc[2]);
        acc[3] = wmma_f16(af, bf3, acc[3]);
    }
    int lane = tid & 31;
    int mrow0 = mbase + wave * 16 + 8 * (lane >> 4);
    int ncol0 = nbase + (lane & 15);
#pragma unroll
    for (int nt = 0; nt < 4; ++nt)
#pragma unroll
        for (int r = 0; r < 8; ++r)
            AXout[(size_t)(mrow0 + r) * XCOLS + ncol0 + nt * 16] = (_Float16)acc[nt][r];
}

// Stage xg = [inp | A@inp | 0-pad] (64 x 160) for one node into LDS, once.
__device__ inline void stage_xg(_Float16* Xg, const _Float16* __restrict__ X,
                                const _Float16* __restrict__ AX, int n, int tid) {
    // ki 0..63: straight vector copies from X (4 per thread, static)
#pragma unroll
    for (int u = 0; u < 4; ++u) {
        int i = tid + u * 128;
        int brow = i >> 3, cc = (i & 7) * 8;
        *(uint4*)&Xg[brow * XGS + cc] =
            *(const uint4*)&X[(size_t)n * XCOLS + brow * IPAD + cc];
    }
    // ki 64..159: boundary region (concat seam + zero pad), elementwise
#pragma unroll 4
    for (int u = 0; u < 48; ++u) {
        int i = tid + u * 128;
        int brow = i / 96, ki = 64 + (i % 96);
        size_t base = (size_t)n * XCOLS + brow * IPAD;
        _Float16 v = (_Float16)0.f;
        if (ki < IGATE)   v = X[base + ki];
        else if (ki < KI) v = AX[base + (ki - IGATE)];
        Xg[brow * XGS + ki] = v;
    }
}

// --------------- per-node zr GEMM + sigmoid + (r*h) fusion -----------------
__global__ void node_zr_kernel(const _Float16* __restrict__ X,
                               const _Float16* __restrict__ AX,
                               const _Float16* __restrict__ Wn,
                               const float* __restrict__ bn,
                               const float* __restrict__ h,
                               float* __restrict__ zbuf,
                               _Float16* __restrict__ X2) {
    __shared__ _Float16 Xg[64 * XGS];
    int tid = threadIdx.x, wave = tid >> 5, n = blockIdx.x;
    const _Float16* Wn_n = Wn + (size_t)n * 128 * KPAD;

    stage_xg(Xg, X, AX, n, tid);
    __syncthreads();

    v8f acc[8] = {};
#pragma unroll
    for (int kc = 0; kc < KPAD / 32; ++kc) {
        v16h af = frag_a(Xg, XGS, wave * 16, kc * 32);
        v16h bf[8];
#pragma unroll
        for (int ot = 0; ot < 8; ++ot) bf[ot] = frag_b_w(Wn_n, ot * 16, kc * 32);
#pragma unroll
        for (int ot = 0; ot < 8; ++ot) acc[ot] = wmma_f16(af, bf[ot], acc[ot]);
    }
    int lane = tid & 31;
    int bq = wave * 16 + 8 * (lane >> 4);
#pragma unroll
    for (int ot = 0; ot < 8; ++ot) {
        int o = ot * 16 + (lane & 15);
        float bias = bn[n * 128 + o];
#pragma unroll
        for (int r = 0; r < 8; ++r) {
            int b = bq + r;
            float s = 1.f / (1.f + __expf(-(acc[ot][r] + bias)));
            if (o < HID) {
                zbuf[((size_t)b * N_NODES + n) * HID + o] = s;             // z
            } else {
                float hv = h[((size_t)b * N_NODES + n) * HID + (o - HID)]; // r*h
                X2[(size_t)n * XCOLS + b * IPAD + CIN + (o - HID)] = (_Float16)(s * hv);
            }
        }
    }
}

// --------------- per-node candidate GEMM + tanh + GRU update ---------------
__global__ void node_c_kernel(const _Float16* __restrict__ X,
                              const _Float16* __restrict__ AX,
                              const _Float16* __restrict__ Wn,
                              const float* __restrict__ bn,
                              const float* __restrict__ zbuf,
                              float* __restrict__ h,
                              _Float16* __restrict__ X1) {
    __shared__ _Float16 Xg[64 * XGS];
    int tid = threadIdx.x, wave = tid >> 5, n = blockIdx.x;
    const _Float16* Wn_n = Wn + (size_t)n * 64 * KPAD;

    stage_xg(Xg, X, AX, n, tid);
    __syncthreads();

    v8f acc[4] = {};
#pragma unroll
    for (int kc = 0; kc < KPAD / 32; ++kc) {
        v16h af = frag_a(Xg, XGS, wave * 16, kc * 32);
        v16h bf[4];
#pragma unroll
        for (int ot = 0; ot < 4; ++ot) bf[ot] = frag_b_w(Wn_n, ot * 16, kc * 32);
#pragma unroll
        for (int ot = 0; ot < 4; ++ot) acc[ot] = wmma_f16(af, bf[ot], acc[ot]);
    }
    int lane = tid & 31;
    int bq = wave * 16 + 8 * (lane >> 4);
#pragma unroll
    for (int ot = 0; ot < 4; ++ot) {
        int o = ot * 16 + (lane & 15);
        float bias = bn[n * HID + o];
#pragma unroll
        for (int r = 0; r < 8; ++r) {
            int b = bq + r;
            float hc = tanhf(acc[ot][r] + bias);
            size_t hidx = ((size_t)b * N_NODES + n) * HID + o;
            float z = zbuf[hidx];
            float hn = z * h[hidx] + (1.f - z) * hc;
            h[hidx] = hn;
            X1[(size_t)n * XCOLS + b * IPAD + CIN + o] = (_Float16)hn;  // next step
        }
    }
}

// ----------------------------- readout -------------------------------------
__global__ void readout_kernel(const float* __restrict__ h,
                               const float* __restrict__ Wout,
                               const float* __restrict__ bout,
                               float* __restrict__ out) {
    int idx = blockIdx.x * blockDim.x + threadIdx.x;  // B*N
    const float* hp = h + (size_t)idx * HID;
    float a = bout[0];
#pragma unroll
    for (int o = 0; o < HID; ++o) a += hp[o] * Wout[o];
    out[idx] = a;
}

// ---------------------------------------------------------------------------
extern "C" void kernel_launch(void* const* d_in, const int* in_sizes, int n_in,
                              void* d_out, int out_size, void* d_ws, size_t ws_size,
                              hipStream_t stream) {
    const float* x     = (const float*)d_in[0];
    const float* E1    = (const float*)d_in[1];
    const float* W_zr  = (const float*)d_in[2];
    const float* b_zr  = (const float*)d_in[3];
    const float* W_c   = (const float*)d_in[4];
    const float* b_c   = (const float*)d_in[5];
    const float* W_out = (const float*)d_in[6];
    const float* b_out = (const float*)d_in[7];
    float* out = (float*)d_out;

    char* p = (char*)d_ws;
    auto alloc = [&](size_t bytes) -> void* {
        void* r = (void*)p;
        p += (bytes + 255) & ~(size_t)255;
        return r;
    };
    _Float16* A16  = (_Float16*)alloc((size_t)N_NODES * N_NODES * 2);
    _Float16* Wnzr = (_Float16*)alloc((size_t)N_NODES * 128 * KPAD * 2);  // (n,o,k)
    _Float16* Wnc  = (_Float16*)alloc((size_t)N_NODES * 64 * KPAD * 2);   // (n,o,k)
    float*    bnzr = (float*)alloc((size_t)N_NODES * 128 * 4);
    float*    bnc  = (float*)alloc((size_t)N_NODES * 64 * 4);
    _Float16* X1   = (_Float16*)alloc((size_t)N_NODES * XCOLS * 2);
    _Float16* X2   = (_Float16*)alloc((size_t)N_NODES * XCOLS * 2);
    _Float16* AX1  = (_Float16*)alloc((size_t)N_NODES * XCOLS * 2);
    _Float16* AX2  = (_Float16*)alloc((size_t)N_NODES * XCOLS * 2);
    float*    zbuf = (float*)alloc((size_t)BATCH * N_NODES * HID * 4);
    float*    hbuf = (float*)alloc((size_t)BATCH * N_NODES * HID * 4);

    hipMemsetAsync(X1, 0, (size_t)N_NODES * XCOLS * 2, stream);
    hipMemsetAsync(X2, 0, (size_t)N_NODES * XCOLS * 2, stream);
    hipMemsetAsync(hbuf, 0, (size_t)BATCH * N_NODES * HID * 4, stream);

    adj_kernel<<<N_NODES, 128, 0, stream>>>(E1, A16);
    nodew_kernel<<<(N_NODES * KPAD * 128) / 256, 256, 0, stream>>>(E1, W_zr, Wnzr, 128);
    nodew_kernel<<<(N_NODES * KPAD * 64) / 256, 256, 0, stream>>>(E1, W_c, Wnc, 64);
    bias_kernel<<<(N_NODES * 192) / 256, 256, 0, stream>>>(E1, b_zr, b_c, bnzr, bnc);

    for (int t = 0; t < TSTEPS; ++t) {
        pack_kernel<<<(BATCH * N_NODES * CIN) / 256, 256, 0, stream>>>(x, t, X1, X2);
        gemmA_kernel<<<dim3(N_NODES / 64, XCOLS / 64), 128, 0, stream>>>(A16, X1, AX1);
        node_zr_kernel<<<N_NODES, 128, 0, stream>>>(X1, AX1, Wnzr, bnzr, hbuf, zbuf, X2);
        gemmA_kernel<<<dim3(N_NODES / 64, XCOLS / 64), 128, 0, stream>>>(A16, X2, AX2);
        node_c_kernel<<<N_NODES, 128, 0, stream>>>(X2, AX2, Wnc, bnc, zbuf, hbuf, X1);
    }

    readout_kernel<<<(BATCH * N_NODES) / 256, 256, 0, stream>>>(hbuf, W_out, b_out, out);
}
